// MailboxAttention_47837345743065
// MI455X (gfx1250) — matrix operand order repaired and checked by reading, compiled, not run on hardware
//
#include <hip/hip_runtime.h>
#include <math.h>

#define DIM 128
#define INV_SQRT_DIM 0.08838834764831845f

typedef __attribute__((ext_vector_type(16))) _Float16 v16h;
typedef __attribute__((ext_vector_type(8)))  float    v8f;

// Monotone int encoding of float: signed-int compare order == float order.
__device__ __forceinline__ int encodeF(float f) {
    int i = __float_as_int(f);
    return (i >= 0) ? i : (i ^ 0x7fffffff);
}
__device__ __forceinline__ float decodeI(int i) {
    int j = (i >= 0) ? i : (i ^ 0x7fffffff);
    return __int_as_float(j);
}

// Native f32 atomic add (global_atomic_add_f32) — coarse-grained device
// memory only, which is all we touch here.
__device__ __forceinline__ void atomAddF(float* p, float v) {
    unsafeAtomicAdd(p, v);
}

// ---------------------------------------------------------------------------
// Kernel 0: zero pooled, init segment-max to -inf (encoded), denom to 0.
// ---------------------------------------------------------------------------
__global__ void mb_init_kernel(float* __restrict__ pooled,
                               int* __restrict__ maxb,
                               float* __restrict__ denom, int A) {
    int t = blockIdx.x * blockDim.x + threadIdx.x;
    if (t < A * DIM) pooled[t] = 0.0f;
    if (t < A) {
        maxb[t]  = encodeF(-INFINITY);
        denom[t] = 0.0f;
    }
}

// ---------------------------------------------------------------------------
// Kernel 1: per-edge scores via WMMA (16 edges per wave32), segmented atomic
// max over recipients. scores[e] = (q_j . k_i) / sqrt(128).
// ---------------------------------------------------------------------------
__global__ __launch_bounds__(256)
void mb_scores_kernel(const float* __restrict__ z,
                      const long long* __restrict__ i_idx,
                      const long long* __restrict__ j_idx,
                      float* __restrict__ scores,
                      int* __restrict__ maxb,
                      int K) {
    const int lane = threadIdx.x & 31;
    const long long gwave = (long long)blockIdx.x * 8 + (threadIdx.x >> 5);
    const long long e0 = gwave * 16;
    if (e0 >= K) return;                       // wave-uniform: EXEC stays full

    const int  sub   = lane & 15;              // which of 16 edges
    const long long eRaw = e0 + sub;
    const bool valid = (eRaw < (long long)K);
    const long long e = valid ? eRaw : (long long)(K - 1);   // clamp tail loads

    const int jrow = (int)j_idx[e];
    const int irow = (int)i_idx[e];
    const float* __restrict__ qrow = z + (long long)jrow * DIM;  // A row (query)
    const float* __restrict__ krow = z + (long long)irow * DIM;  // B col (key)

    const int hi = lane >> 4;                  // which half-wave
    v8f acc = {};

    #pragma unroll
    for (int c0 = 0; c0 < DIM; c0 += 32) {
        // A fragment (16x32 f16): lanes 0-15 K=c0+0..7 & c0+16..23; hi-half +8
        const int aBase = c0 + (hi << 3);
        const float4 a0 = *(const float4*)(qrow + aBase);
        const float4 a1 = *(const float4*)(qrow + aBase + 4);
        const float4 a2 = *(const float4*)(qrow + aBase + 16);
        const float4 a3 = *(const float4*)(qrow + aBase + 20);
        // B fragment (32x16 f16): lanes 0-15 K=c0+0..7 & c0+16..23; hi-half +8
        const int bBase = c0 + (hi << 3);
        const float4 b0 = *(const float4*)(krow + bBase);
        const float4 b1 = *(const float4*)(krow + bBase + 4);
        const float4 b2 = *(const float4*)(krow + bBase + 16);
        const float4 b3 = *(const float4*)(krow + bBase + 20);

        v16h af, bf;
        af[0]  = (_Float16)a0.x; af[1]  = (_Float16)a0.y;
        af[2]  = (_Float16)a0.z; af[3]  = (_Float16)a0.w;
        af[4]  = (_Float16)a1.x; af[5]  = (_Float16)a1.y;
        af[6]  = (_Float16)a1.z; af[7]  = (_Float16)a1.w;
        af[8]  = (_Float16)a2.x; af[9]  = (_Float16)a2.y;
        af[10] = (_Float16)a2.z; af[11] = (_Float16)a2.w;
        af[12] = (_Float16)a3.x; af[13] = (_Float16)a3.y;
        af[14] = (_Float16)a3.z; af[15] = (_Float16)a3.w;

        bf[0]  = (_Float16)b0.x; bf[1]  = (_Float16)b0.y;
        bf[2]  = (_Float16)b0.z; bf[3]  = (_Float16)b0.w;
        bf[4]  = (_Float16)b1.x; bf[5]  = (_Float16)b1.y;
        bf[6]  = (_Float16)b1.z; bf[7]  = (_Float16)b1.w;
        bf[8]  = (_Float16)b2.x; bf[9]  = (_Float16)b2.y;
        bf[10] = (_Float16)b2.z; bf[11] = (_Float16)b2.w;
        bf[12] = (_Float16)b3.x; bf[13] = (_Float16)b3.y;
        bf[14] = (_Float16)b3.z; bf[15] = (_Float16)b3.w;

        acc = __builtin_amdgcn_wmma_f32_16x16x32_f16(
            false, af, false, bf, (short)0, acc, false, false);
    }

    // Diagonal extraction: S(m,m) sits at acc[m] lane m (m<8), acc[m-8] lane m+16.
    float dv = 0.0f;
    #pragma unroll
    for (int r = 0; r < 8; ++r)
        if (lane == r || lane == r + 24) dv = acc[r];
    // Compact diag into lanes 0..15.
    const float sc = __shfl(dv, (lane < 8) ? lane : (lane + 16), 32);

    if (lane < 16 && valid) {
        const float s = sc * INV_SQRT_DIM;
        scores[eRaw] = s;
        atomicMax(&maxb[jrow], encodeF(s));
    }
}

// ---------------------------------------------------------------------------
// Kernel 2: exp(score - max[j]); atomicAdd denom[j]; unnormalized weighted
// scatter pooled[j] += exp * k_i   (one wave32 per edge; lane handles 4 dims)
// ---------------------------------------------------------------------------
__global__ __launch_bounds__(256)
void mb_scatter_kernel(const float* __restrict__ z,
                       const long long* __restrict__ i_idx,
                       const long long* __restrict__ j_idx,
                       const float* __restrict__ scores,
                       const int* __restrict__ maxb,
                       float* __restrict__ denom,
                       float* __restrict__ pooled,
                       int K) {
    const int lane = threadIdx.x & 31;
    const long long e = (long long)blockIdx.x * 8 + (threadIdx.x >> 5);
    if (e >= K) return;

    const int j = (int)j_idx[e];
    const int i = (int)i_idx[e];
    const float m  = decodeI(maxb[j]);
    const float ev = __expf(scores[e] - m);
    if (lane == 0) atomAddF(&denom[j], ev);

    const float4 kv = *(const float4*)(z + (long long)i * DIM + lane * 4);
    float* p = pooled + (long long)j * DIM + lane * 4;
    atomAddF(p + 0, ev * kv.x);
    atomAddF(p + 1, ev * kv.y);
    atomAddF(p + 2, ev * kv.z);
    atomAddF(p + 3, ev * kv.w);
}

// ---------------------------------------------------------------------------
// Kernel 3: pooled[a, :] /= (denom[a] + 1e-6)
// ---------------------------------------------------------------------------
__global__ void mb_finalize_kernel(float* __restrict__ pooled,
                                   const float* __restrict__ denom, int A) {
    int t = blockIdx.x * blockDim.x + threadIdx.x;
    if (t < A * DIM) {
        const int a = t >> 7;
        pooled[t] = pooled[t] / (denom[a] + 1e-6f);
    }
}

extern "C" void kernel_launch(void* const* d_in, const int* in_sizes, int n_in,
                              void* d_out, int out_size, void* d_ws, size_t ws_size,
                              hipStream_t stream) {
    const float* z           = (const float*)d_in[0];
    // d_in[1] (recv_query_all) unused by the reference forward.
    const long long* i_idx   = (const long long*)d_in[2];
    const long long* j_idx   = (const long long*)d_in[3];
    const int K = in_sizes[2];
    const int A = in_sizes[0] / DIM;

    float* pooled = (float*)d_out;
    char*  ws     = (char*)d_ws;
    float* scores = (float*)ws;                                   // K floats
    int*   maxb   = (int*)(ws + (size_t)K * 4);                   // A ints
    float* denom  = (float*)(ws + (size_t)K * 4 + (size_t)A * 4); // A floats

    const int initN = A * DIM;
    mb_init_kernel<<<(initN + 255) / 256, 256, 0, stream>>>(pooled, maxb, denom, A);

    const long long waves = ((long long)K + 15) / 16;
    const int blocks1 = (int)((waves + 7) / 8);
    mb_scores_kernel<<<blocks1, 256, 0, stream>>>(z, i_idx, j_idx, scores, maxb, K);

    const int blocks2 = (K + 7) / 8;
    mb_scatter_kernel<<<blocks2, 256, 0, stream>>>(z, i_idx, j_idx, scores, maxb,
                                                   denom, pooled, K);

    mb_finalize_kernel<<<(initN + 255) / 256, 256, 0, stream>>>(pooled, denom, A);
}
